// GCN_83820581749458
// MI455X (gfx1250) — matrix operand order repaired and checked by reading, not compiled, not run on hardware
//
#include <hip/hip_runtime.h>
#include <hip/hip_bf16.h>

// ---------------------------------------------------------------------------
// GCN forward for MI455X (gfx1250, wave32, WMMA).
//   ys = elu(adj @ (elu(zs) @ W0) + b0);  ys = elu(adj @ (ys @ W1) + b1)
//   (same for zu), out = [ys|yu] @ Wl^T + bl
// Two adj-GEMM passes over both branches at once (N x 128), f16 operands /
// f32 accumulation via v_wmma_f32_16x16x32_f16. adj read exactly twice from
// HBM (memory-bound optimum, ~92us floor at 23.3 TB/s). Double-buffered LDS
// B panel (1 barrier / K-step), B tiles in distinct registers so ds_load
// latency overlaps WMMA. Dense epilogues fused via WMMA.
// ---------------------------------------------------------------------------

#define NN 16384
#define DIN 64

typedef __attribute__((ext_vector_type(16))) _Float16 v16h;
typedef __attribute__((ext_vector_type(8)))  _Float16 v8h;
typedef __attribute__((ext_vector_type(2)))  __fp16   v2fp;   // cvt_pkrtz result
typedef __attribute__((ext_vector_type(8)))  float    v8f;
typedef __attribute__((ext_vector_type(4)))  float    v4f;

__device__ __forceinline__ float eluf(float x) {
    return x > 0.f ? x : (__expf(x) - 1.f);
}

__device__ __forceinline__ v8f wmma16(v16h a, v16h b, v8f c) {
    return __builtin_amdgcn_wmma_f32_16x16x32_f16(
        /*neg_a=*/false, a, /*neg_b=*/false, b,
        /*c_mod=*/(short)0, c, /*reuse_a=*/false, /*reuse_b=*/false);
}

__device__ __forceinline__ v16h mk16(v8h lo, v8h hi) {
    v16h r;
#pragma unroll
    for (int i = 0; i < 8; ++i) { r[i] = lo[i]; r[i + 8] = hi[i]; }
    return r;
}

// Pack 16 f32 (4 x float4, already in A-tile K order) into v16h via
// v_cvt_pk_rtz_f16_f32.
__device__ __forceinline__ v16h packA(v4f a0, v4f a1, v4f a2, v4f a3) {
    v16h r; v2fp q;
    q = __builtin_amdgcn_cvt_pkrtz(a0.x, a0.y); r[0]  = (_Float16)q.x; r[1]  = (_Float16)q.y;
    q = __builtin_amdgcn_cvt_pkrtz(a0.z, a0.w); r[2]  = (_Float16)q.x; r[3]  = (_Float16)q.y;
    q = __builtin_amdgcn_cvt_pkrtz(a1.x, a1.y); r[4]  = (_Float16)q.x; r[5]  = (_Float16)q.y;
    q = __builtin_amdgcn_cvt_pkrtz(a1.z, a1.w); r[6]  = (_Float16)q.x; r[7]  = (_Float16)q.y;
    q = __builtin_amdgcn_cvt_pkrtz(a2.x, a2.y); r[8]  = (_Float16)q.x; r[9]  = (_Float16)q.y;
    q = __builtin_amdgcn_cvt_pkrtz(a2.z, a2.w); r[10] = (_Float16)q.x; r[11] = (_Float16)q.y;
    q = __builtin_amdgcn_cvt_pkrtz(a3.x, a3.y); r[12] = (_Float16)q.x; r[13] = (_Float16)q.y;
    q = __builtin_amdgcn_cvt_pkrtz(a3.z, a3.w); r[14] = (_Float16)q.x; r[15] = (_Float16)q.y;
    return r;
}

// ---------------------------------------------------------------------------
// Prep: XT0[c][r] = (f16)( elu(z[r, branch-half]) @ W0 ), c in [0,128),
// cols 0..63 = s-branch, 64..127 = u-branch (stored transposed for B-tiles).
// Also builds f16 weight matrices in WMMA B layout:
//   Wc1[j*64 + k]  = (f16) W1[k][j]      (64x64)
//   Wl16[o*128+ c] = (f16) Wl[o][c]      (32x128)
// ---------------------------------------------------------------------------
__global__ __launch_bounds__(256)
void gcn_prep(const float* __restrict__ z, const float* __restrict__ W0,
              const float* __restrict__ W1, const float* __restrict__ Wl,
              _Float16* __restrict__ XT0, _Float16* __restrict__ Wc1,
              _Float16* __restrict__ Wl16) {
    __shared__ float sW0[DIN * DIN];
    const int tid = threadIdx.x;
    for (int i = tid; i < DIN * DIN; i += 256) sW0[i] = W0[i];
    if (blockIdx.x == 0) {
        for (int i = tid; i < DIN * DIN; i += 256) {
            int k = i >> 6, j = i & 63;
            Wc1[j * DIN + k] = (_Float16)W1[i];
        }
        for (int i = tid; i < 32 * 128; i += 256) Wl16[i] = (_Float16)Wl[i];
    }
    __syncthreads();

    const int r = blockIdx.x * 256 + tid;
    const float* zr = z + (size_t)r * (2 * DIN);
    float es[DIN], eu[DIN];
#pragma unroll
    for (int i = 0; i < DIN; ++i) {
        es[i] = eluf(zr[i]);
        eu[i] = eluf(zr[DIN + i]);
    }
    for (int c = 0; c < DIN; ++c) {
        float s = 0.f, u = 0.f;
#pragma unroll
        for (int i = 0; i < DIN; ++i) {
            float w = sW0[i * DIN + c];
            s += es[i] * w;
            u += eu[i] * w;
        }
        XT0[(size_t)c * NN + r]         = (_Float16)s;   // coalesced across lanes
        XT0[(size_t)(DIN + c) * NN + r] = (_Float16)u;
    }
}

// ---------------------------------------------------------------------------
// Main GEMM: Y = elu(adj @ X + bias[col%64]) for 128 cols (both branches).
// MODE 0: X1 = Y @ W1 (per branch, via Wc1), written transposed f16 -> XTout.
// MODE 1: out = Y @ Wl^T + bl, written fp32 -> out.
// Block: 256 threads / 8 waves; block tile = 128 rows x 128 cols;
// wave w owns row-tile w (16 rows) x all 8 col-tiles. K loop: 32/step,
// double-buffered LDS B panel, unrolled x2 for buffer parity.
// ---------------------------------------------------------------------------
template <int MODE>
__global__ __launch_bounds__(256)
void gcn_gemm(const float* __restrict__ adj, const _Float16* __restrict__ XTin,
              const float* __restrict__ bias, const _Float16* __restrict__ Wn,
              const float* __restrict__ bias2, _Float16* __restrict__ XTout,
              float* __restrict__ out) {
    // LDS: 2 x B staging panel [128 cols][40 halves], epilogue H [128][152]
    __shared__ _Float16 smem[2 * 128 * 40 + 128 * 152];
    _Float16* Bs0 = smem;
    _Float16* Bs1 = smem + 128 * 40;
    _Float16* Hs  = smem + 2 * 128 * 40;

    const int tid  = threadIdx.x;
    const int wave = tid >> 5;
    const int lane = tid & 31;
    const int l16  = lane & 15;
    const int hi   = lane >> 4;   // 0: lanes 0-15, 1: lanes 16-31

    const int rowBlk = blockIdx.x * 128;
    const float* aPtr =
        adj + (size_t)(rowBlk + wave * 16 + l16) * NN + hi * 8;

    // B staging: thread covers (col sc, 16-half chunk sh)
    const int sc = tid >> 1;
    const int sh = tid & 1;
    const _Float16* sPtr = XTin + (size_t)sc * NN + sh * 16;
    const int sOff = sc * 40 + sh * 16;

    v8f acc[8];
#pragma unroll
    for (int i = 0; i < 8; ++i) acc[i] = v8f{};

    // Prologue: stage k=0 panel into Bs0, prefetch k=32 into registers.
    v8h p0 = *(const v8h*)(sPtr);
    v8h p1 = *(const v8h*)(sPtr + 8);
    *(v8h*)(Bs0 + sOff)     = p0;
    *(v8h*)(Bs0 + sOff + 8) = p1;
    p0 = *(const v8h*)(sPtr + 32);
    p1 = *(const v8h*)(sPtr + 40);

    // One K-step: read rdP (data kA), store p0/p1 (data kA+32) into wrP,
    // prefetch data kPre into p0/p1. Single barrier per step.
    auto kstep = [&](const _Float16* rdP, _Float16* wrP, int kA, int kPre) {
        __syncthreads();   // wrP reads (prev step) done; rdP stores visible
        *(v8h*)(wrP + sOff)     = p0;
        *(v8h*)(wrP + sOff + 8) = p1;

        // A tile (16x32 of adj) straight from HBM, per-lane 4 x b128
        const float* ap = aPtr + kA;
        v4f a0 = *(const v4f*)(ap);
        v4f a1 = *(const v4f*)(ap + 4);
        v4f a2 = *(const v4f*)(ap + 16);
        v4f a3 = *(const v4f*)(ap + 20);
        __builtin_prefetch(ap + 512, 0, 0);   // stream adj row ahead

        if (kPre < NN) {                      // prefetch next B panel
            p0 = *(const v8h*)(sPtr + kPre);
            p1 = *(const v8h*)(sPtr + kPre + 8);
        }
        v16h va = packA(a0, a1, a2, a3);

        // All 8 B tiles into distinct registers: ds_load latency overlaps WMMA
        v16h vb[8];
#pragma unroll
        for (int ct = 0; ct < 8; ++ct) {
            const _Float16* bp = rdP + (16 * ct + l16) * 40 + hi * 8;
            vb[ct] = mk16(*(const v8h*)(bp), *(const v8h*)(bp + 16));
        }
#pragma unroll
        for (int ct = 0; ct < 8; ++ct)
            acc[ct] = wmma16(va, vb[ct], acc[ct]);
    };

    for (int k0 = 0; k0 < NN; k0 += 64) {
        kstep(Bs0, Bs1, k0,      k0 + 64);   // consume k0,    stage k0+32
        kstep(Bs1, Bs0, k0 + 32, k0 + 96);   // consume k0+32, stage k0+64
    }

    // ---------------- epilogue: H = elu(acc + bias), f16 -> LDS ------------
    __syncthreads();
    const int rb = wave * 16;
#pragma unroll
    for (int ct = 0; ct < 8; ++ct) {
        const int col  = 16 * ct + l16;
        const float bv = bias[col & 63];
        const int hoff = (MODE == 0) ? ((col < 64) ? col : col + 16) : col;
#pragma unroll
        for (int v = 0; v < 8; ++v) {
            float y = acc[ct][v] + bv;
            Hs[(rb + v + hi * 8) * 152 + hoff] = (_Float16)eluf(y);
        }
    }
    __syncthreads();

    if (MODE == 0) {
        // X1 = H @ W1 per branch (K = 64, 2 WMMA k-steps), cols 0..127
        v8f xacc[8];
#pragma unroll
        for (int i = 0; i < 8; ++i) xacc[i] = v8f{};
#pragma unroll
        for (int kk = 0; kk < 2; ++kk) {
#pragma unroll
            for (int b = 0; b < 2; ++b) {
                const _Float16* hp =
                    Hs + (rb + l16) * 152 + b * 80 + kk * 32 + hi * 8;
                v16h va = mk16(*(const v8h*)(hp), *(const v8h*)(hp + 16));
#pragma unroll
                for (int ctb = 0; ctb < 4; ++ctb) {
                    const _Float16* wp =
                        Wn + (16 * ctb + l16) * 64 + kk * 32 + hi * 8;
                    v16h vb = mk16(*(const v8h*)(wp), *(const v8h*)(wp + 16));
                    xacc[b * 4 + ctb] = wmma16(va, vb, xacc[b * 4 + ctb]);
                }
            }
        }
#pragma unroll
        for (int ct = 0; ct < 8; ++ct) {
            const int col = 16 * ct + l16;
#pragma unroll
            for (int v = 0; v < 8; ++v) {
                const int row = rowBlk + rb + v + hi * 8;
                XTout[(size_t)col * NN + row] = (_Float16)xacc[ct][v];
            }
        }
    } else {
        // out = Y @ Wl^T + bl   (K = 128, 4 WMMA k-steps, 32 output cols)
        v8f oacc[2];
        oacc[0] = v8f{}; oacc[1] = v8f{};
#pragma unroll
        for (int kk = 0; kk < 4; ++kk) {
            const _Float16* hp = Hs + (rb + l16) * 152 + kk * 32 + hi * 8;
            v16h va = mk16(*(const v8h*)(hp), *(const v8h*)(hp + 16));
#pragma unroll
            for (int ct = 0; ct < 2; ++ct) {
                const _Float16* wp =
                    Wn + (16 * ct + l16) * 128 + kk * 32 + hi * 8;
                v16h vb = mk16(*(const v8h*)(wp), *(const v8h*)(wp + 16));
                oacc[ct] = wmma16(va, vb, oacc[ct]);
            }
        }
#pragma unroll
        for (int ct = 0; ct < 2; ++ct) {
            const int col = 16 * ct + l16;
            const float bo = bias2[col];
#pragma unroll
            for (int v = 0; v < 8; ++v) {
                const int row = rowBlk + rb + v + hi * 8;
                out[(size_t)row * 32 + col] = oacc[ct][v] + bo;
            }
        }
    }
}

// ---------------------------------------------------------------------------
extern "C" void kernel_launch(void* const* d_in, const int* in_sizes, int n_in,
                              void* d_out, int out_size, void* d_ws,
                              size_t ws_size, hipStream_t stream) {
    const float* z   = (const float*)d_in[0];
    const float* adj = (const float*)d_in[1];
    const float* W0  = (const float*)d_in[2];
    const float* b0  = (const float*)d_in[3];
    const float* W1  = (const float*)d_in[4];
    const float* b1  = (const float*)d_in[5];
    const float* Wl  = (const float*)d_in[6];
    const float* bl  = (const float*)d_in[7];
    float* outp = (float*)d_out;

    char* ws = (char*)d_ws;
    _Float16* XT0  = (_Float16*)(ws);                            // 4 MiB
    _Float16* XT1  = (_Float16*)(ws + (size_t)4 * 1024 * 1024);  // 4 MiB
    _Float16* Wc1  = (_Float16*)(ws + (size_t)8 * 1024 * 1024);  // 8 KiB
    _Float16* Wl16 = (_Float16*)(ws + (size_t)8 * 1024 * 1024 + 16384);

    gcn_prep<<<NN / 256, 256, 0, stream>>>(z, W0, W1, Wl, XT0, Wc1, Wl16);
    gcn_gemm<0><<<NN / 128, 256, 0, stream>>>(adj, XT0, b0, Wc1, nullptr,
                                              XT1, nullptr);
    gcn_gemm<1><<<NN / 128, 256, 0, stream>>>(adj, XT1, b1, Wl16, bl,
                                              nullptr, outp);
}
